// MultiHeadAttention_87703232184622
// MI455X (gfx1250) — matrix operand order repaired and checked by reading, not compiled, not run on hardware
//
#include <hip/hip_runtime.h>
#include <cstdint>
#include <cstddef>

#define BS  16
#define DA  512
#define SL  1024
#define H   8
#define DH  64
#define NEG_INF (-1e9f)

typedef __bf16 bf16;
typedef __attribute__((ext_vector_type(16))) __bf16 v16bf;
typedef __attribute__((ext_vector_type(8)))  __bf16 v8bf;
typedef __attribute__((ext_vector_type(4)))  __bf16 v4bf;
typedef __attribute__((ext_vector_type(8)))  float   v8f;

typedef uint32_t u32x4 __attribute__((ext_vector_type(4)));
typedef int      i32x4 __attribute__((ext_vector_type(4)));
typedef int      i32x8 __attribute__((ext_vector_type(8)));

// ---- Tensor Data Mover availability / arity detection (per CDNA5_HIP.md) ----
#if defined(__has_builtin)
# if __has_builtin(__builtin_amdgcn_tensor_load_to_lds)
#  if __has_include(<hip/amd_detail/amd_gfx1250_TDM.h>)
#   define TDM_MODE 3   // therock-10.0 headers -> 6-arg builtin
#  else
#   define TDM_MODE 2   // ROCm 7.x clang-22 -> 5-arg builtin
#  endif
# endif
#endif
#ifndef TDM_MODE
# define TDM_MODE 0
#endif

union V16U { v16bf v; v8bf h[2]; };

static __device__ __forceinline__ v8f wmma_bf16(v16bf a, v16bf b, v8f c) {
  // D = A(16x32 bf16) * B(32x16 bf16) + C(16x16 f32)
  return __builtin_amdgcn_wmma_f32_16x16x32_bf16(false, a, false, b, (short)0, c, false, false);
}

#if TDM_MODE
// Flat LDS address: low 32 bits are the LDS byte offset (ISA 10.2 aperture rule).
static __device__ __forceinline__ uint32_t lds_off_of(const void* p) {
  return (uint32_t)(uintptr_t)p;
}

// Issue one TDM 2D tile copy global->LDS.
//   tile_w  : contiguous elements per row (bf16 elements)
//   tile_h  : number of rows
//   stride  : row stride in the global tensor (elements)
//   pad_int : pad_interval code (2^(c+1) DWORDs between pads)
//   pad_amt : pad_amount code (c+1 DWORDs inserted)
static __device__ __forceinline__ void tdm_load_2d(const void* gptr, uint32_t lds_off,
                                                   uint32_t tile_w, uint32_t tile_h,
                                                   uint32_t stride,
                                                   uint32_t pad_int, uint32_t pad_amt) {
  uint64_t ga = (uint64_t)(uintptr_t)gptr;
  u32x4 g0 = { 1u,                                   // count=1 (valid user D#)
               lds_off,                              // lds_addr
               (uint32_t)ga,                         // global_addr[31:0]
               (uint32_t)(ga >> 32) | (2u << 30) };  // global_addr[56:32] | type=2
  const uint32_t td = 0x40000000u;                   // huge tensor dims: no OOB clip
  i32x8 g1 = { (int)((1u << 16)                      // data_size = 2 bytes
                   | (1u << 20)                      // pad_enable
                   | (pad_int << 22) | (pad_amt << 25)),
               (int)((td & 0xFFFFu) << 16),          // tensor_dim0[15:0]
               (int)((td >> 16) | ((td & 0xFFFFu) << 16)),
               (int)((td >> 16) | (tile_w << 16)),   // tile_dim0
               (int)tile_h,                          // tile_dim1 (tile_dim2=0)
               (int)stride,                          // tensor_dim0_stride[31:0]
               0, 0 };
  i32x4 g2 = { 1, 1, 0, 0 };                         // tensor_dim2=1, tensor_dim3=1
  i32x4 g3 = { 0, 1 << 16, 0, 0 };                   // tensor_dim4=1
#if TDM_MODE == 3
  i32x8 g4 = { 0, 0, 0, 0, 0, 0, 0, 0 };
  __builtin_amdgcn_tensor_load_to_lds(g0, g1, g2, g3, g4, 0);
#else
  __builtin_amdgcn_tensor_load_to_lds(g0, g1, g2, g3, 0);
#endif
}
#endif  // TDM_MODE

// ---------------------------------------------------------------------------
// Elementwise f32 -> bf16 (weights; 4 elements / thread, vectorized)
// ---------------------------------------------------------------------------
__global__ void cvt_f32_bf16(const float* __restrict__ src, bf16* __restrict__ dst, int n4) {
  int i = blockIdx.x * blockDim.x + threadIdx.x;
  if (i >= n4) return;
  const float4 f = ((const float4*)src)[i];
  v4bf o; o[0] = (bf16)f.x; o[1] = (bf16)f.y; o[2] = (bf16)f.z; o[3] = (bf16)f.w;
  *((v4bf*)dst + i) = o;
}

// ---------------------------------------------------------------------------
// Pack Q: q[b, h*64+d, s] (f32) -> qp[((b*H+h)*SL+s)*64 + d] (bf16)
// ---------------------------------------------------------------------------
__global__ void pack_q(const float* __restrict__ q, bf16* __restrict__ qp) {
  __shared__ float tile[16][17];
  int b = blockIdx.z / H, h = blockIdx.z % H;
  int tx = threadIdx.x, ty = threadIdx.y;
  int d  = blockIdx.y * 16 + ty, s = blockIdx.x * 16 + tx;
  tile[ty][tx] = q[(size_t)(b * DA + h * DH + d) * SL + s];
  __syncthreads();
  int d2 = blockIdx.y * 16 + tx, s2 = blockIdx.x * 16 + ty;
  qp[((size_t)(b * H + h) * SL + s2) * DH + d2] = (bf16)tile[tx][ty];
}

// ---------------------------------------------------------------------------
// Pack X: x[b, c, s] (f32) -> xt[b, s, c] (bf16)  -- transposed activations so
// the projection B-operand tile is a plain 2D strided tile (TDM-friendly).
// ---------------------------------------------------------------------------
__global__ void pack_x(const float* __restrict__ x, bf16* __restrict__ xt) {
  __shared__ float tile[16][17];
  int b = blockIdx.z;
  int tx = threadIdx.x, ty = threadIdx.y;
  int c = blockIdx.y * 16 + ty, s = blockIdx.x * 16 + tx;
  tile[ty][tx] = x[(size_t)(b * DA + c) * SL + s];
  __syncthreads();
  int c2 = blockIdx.y * 16 + tx, s2 = blockIdx.x * 16 + ty;
  xt[((size_t)b * SL + s2) * DA + c2] = (bf16)tile[tx][ty];
}

// ---------------------------------------------------------------------------
// Projection GEMM:  C[cout, s] = W[cout, cin] * X^T[b, s, cin] + bias[cout]
// Both A (weights) and B (activations) tiles TDM-staged, double-buffered.
// isV == 0 : K written as kout[b, h, s, d]   (per-position rows, 16B packed)
// isV == 1 : V written as vout[b, c, s]      (channel rows)
// ---------------------------------------------------------------------------
__global__ __launch_bounds__(256)
void project_gemm(const bf16* __restrict__ wA, const float* __restrict__ bias,
                  const bf16* __restrict__ xT, bf16* __restrict__ kout,
                  bf16* __restrict__ vout, int isV) {
  __shared__ bf16 ldsA[2][128][40];   // [buf][m][cin-tile], rows padded 64B->80B
  __shared__ bf16 ldsBt[2][64][48];   // [buf][n][cin-tile], rows padded 64B->96B

  int b  = blockIdx.z;
  int n0 = blockIdx.x * 64, m0 = blockIdx.y * 128;
  int t = threadIdx.x, lane = t & 31, w = t >> 5;
  int wm = w & 3, wn = w >> 2, half = lane >> 4, lrow = lane & 15;

  const bf16* abase = wA + (size_t)m0 * DA;                 // rows: cout, stride DA
  const bf16* bbase = xT + ((size_t)b * SL + n0) * DA;      // rows: s,    stride DA

  v8f acc[2][2] = {};
  const int NS = DA / 32;

#if !TDM_MODE
  auto stage_copy = [&](int ks2, int bu) {
    int kk = ks2 * 32;
#pragma unroll
    for (int i = 0; i < 2; ++i) {                 // A: 128 rows x 32
      int cA = t + i * 256;
      int mr = cA >> 2, c0 = (cA & 3) * 8;
      v8bf a = *(const v8bf*)(abase + (size_t)mr * DA + kk + c0);
      *(v8bf*)&ldsA[bu][mr][c0] = a;
    }
    { int row = t >> 2, c0 = (t & 3) * 8;         // B: 64 rows x 32, direct copy
      v8bf bv = *(const v8bf*)(bbase + (size_t)row * DA + kk + c0);
      *(v8bf*)&ldsBt[bu][row][c0] = bv; }
  };
#endif

  // Prologue: stage k-step 0 into buffer 0
#if TDM_MODE
  if (w == 0) {
    tdm_load_2d(abase, lds_off_of(&ldsA[0][0][0]),  32, 128, DA, /*16DW*/3, /*4DW*/3);
    tdm_load_2d(bbase, lds_off_of(&ldsBt[0][0][0]), 32, 64,  DA, /*16DW*/3, /*8DW*/7);
  }
#else
  stage_copy(0, 0);
#endif

  for (int ks = 0; ks < NS; ++ks) {
    int buf = ks & 1;
    __syncthreads();                    // all waves done with buf[(ks+1)&1]
#if TDM_MODE
    if (w == 0) {
      if (ks + 1 < NS) {
        int nb = (ks + 1) & 1;
        int kk = (ks + 1) * 32;
        tdm_load_2d(abase + kk, lds_off_of(&ldsA[nb][0][0]),  32, 128, DA, 3, 3);
        tdm_load_2d(bbase + kk, lds_off_of(&ldsBt[nb][0][0]), 32, 64,  DA, 3, 7);
        __builtin_amdgcn_s_wait_tensorcnt(2);   // step ks done; ks+1 in flight
      } else {
        __builtin_amdgcn_s_wait_tensorcnt(0);
      }
    }
#else
    if (ks + 1 < NS) stage_copy(ks + 1, (ks + 1) & 1);
#endif
    __syncthreads();                    // buf[ks&1] ready

    v16bf af[2], bfr[2];
#pragma unroll
    for (int mt = 0; mt < 2; ++mt) {
      int mr = wm * 32 + mt * 16 + lrow;
      V16U u;
      u.h[0] = *(const v8bf*)&ldsA[buf][mr][16 * half];       // K = 0..7  / 16..23
      u.h[1] = *(const v8bf*)&ldsA[buf][mr][16 * half + 8];   // K = 8..15 / 24..31
      af[mt] = u.v;
    }
#pragma unroll
    for (int nt = 0; nt < 2; ++nt) {
      int nc = wn * 32 + nt * 16 + lrow;
      bfr[nt] = *(const v16bf*)&ldsBt[buf][nc][16 * half];    // K = 0..15 / 16..31
    }
#pragma unroll
    for (int mt = 0; mt < 2; ++mt)
#pragma unroll
      for (int nt = 0; nt < 2; ++nt)
        acc[mt][nt] = wmma_bf16(af[mt], bfr[nt], acc[mt][nt]);
  }

  // Epilogue: bias + store
#pragma unroll
  for (int mt = 0; mt < 2; ++mt) {
    int cb = m0 + wm * 32 + mt * 16 + 8 * half;          // cout for r=0 (mult of 8)
    float4 b0 = *(const float4*)(bias + cb);
    float4 b1 = *(const float4*)(bias + cb + 4);
    float bv[8] = {b0.x, b0.y, b0.z, b0.w, b1.x, b1.y, b1.z, b1.w};
#pragma unroll
    for (int nt = 0; nt < 2; ++nt) {
      int s = n0 + wn * 32 + nt * 16 + lrow;
      if (!isV) {                                        // K -> [b,h,s,d], 16B packed
        int hh = cb / DH, dv0 = cb % DH;
        v8bf pv;
#pragma unroll
        for (int r = 0; r < 8; ++r) pv[r] = (bf16)(acc[mt][nt][r] + bv[r]);
        *(v8bf*)(kout + ((size_t)(b * H + hh) * SL + s) * DH + dv0) = pv;
      } else {                                           // V -> [b,c,s]
#pragma unroll
        for (int r = 0; r < 8; ++r)
          vout[((size_t)b * DA + cb + r) * SL + s] = (bf16)(acc[mt][nt][r] + bv[r]);
      }
    }
  }
}

// ---------------------------------------------------------------------------
// Fused flash attention with GBN affine + masking.
// Workgroup = (b, h, 128 q rows); 8 waves x 16 rows; 32-wide key tiles.
// K/V tiles double-buffered; staged by TDM (tensor_load_to_lds) when available.
// ---------------------------------------------------------------------------
__global__ __launch_bounds__(256)
void attention(const bf16* __restrict__ qp, const bf16* __restrict__ kp,
               const bf16* __restrict__ vbf, const int* __restrict__ mask,
               const float* __restrict__ gg, const float* __restrict__ gb,
               const float* __restrict__ gm, const float* __restrict__ gs,
               float* __restrict__ out) {
  __shared__ bf16 ldsK[2][32][80];    // [buf][ks][d], rows padded 128B->160B
  __shared__ bf16 ldsV[2][64][48];    // [buf][dv][ks], rows padded 64B->96B
  __shared__ bf16 ldsP[8][16][48];    // per-wave probability tile

  int b = blockIdx.z, h = blockIdx.y;
  int qrow0 = blockIdx.x * 128;
  int t = threadIdx.x, lane = t & 31, w = t >> 5;
  int half = lane >> 4, lrow = lane & 15;
  int qw = qrow0 + w * 16;

  float alpha = gg[h] / gs[h];
  float beta  = gb[h] - gm[h] * alpha;

  const bf16* kbase = kp + ((size_t)(b * H + h) * SL) * DH;   // rows: s, width d
  const bf16* vbase = vbf + ((size_t)b * DA + h * DH) * SL;   // rows: dv (stride SL)

  // Q A-fragments (d 0..31 and 32..63), loaded once
  const bf16* qrow = qp + ((size_t)(b * H + h) * SL + qw + lrow) * DH;
  V16U ua, ub;
  ua.h[0] = *(const v8bf*)(qrow + 16 * half);
  ua.h[1] = *(const v8bf*)(qrow + 8 + 16 * half);
  ub.h[0] = *(const v8bf*)(qrow + 32 + 16 * half);
  ub.h[1] = *(const v8bf*)(qrow + 40 + 16 * half);
  v16bf qa = ua.v, qb = ub.v;

  // Query-row masks for the 8 rows this lane's C-registers cover
  int4 mq0 = *(const int4*)(mask + b * SL + qw + 8 * half);
  int4 mq1 = *(const int4*)(mask + b * SL + qw + 8 * half + 4);
  int rmask[8] = {mq0.x, mq0.y, mq0.z, mq0.w, mq1.x, mq1.y, mq1.z, mq1.w};

  float m8[8], l8[8];
  v8f o[4] = {};
#pragma unroll
  for (int r = 0; r < 8; ++r) { m8[r] = -1e30f; l8[r] = 0.f; }

  const int NT = SL / 32;

#if !TDM_MODE
  auto stage_copy = [&](int kt2, int bu) {
    { int row = t >> 3, c0 = (t & 7) * 8;   // K: 32 rows x 64, direct copy
      v8bf kv = *(const v8bf*)(kbase + (size_t)(kt2 * 32 + row) * DH + c0);
      *(v8bf*)&ldsK[bu][row][c0] = kv; }
    { int row = t >> 2, c0 = (t & 3) * 8;   // V: 64 rows x 32, direct copy
      v8bf vv = *(const v8bf*)(vbase + (size_t)row * SL + kt2 * 32 + c0);
      *(v8bf*)&ldsV[bu][row][c0] = vv; }
  };
#endif

  // Prologue: stage tile 0 into buffer 0
#if TDM_MODE
  if (w == 0) {
    tdm_load_2d(kbase, lds_off_of(&ldsK[0][0][0]), 64, 32, DH, /*32DW*/4, /*8DW*/7);
    tdm_load_2d(vbase, lds_off_of(&ldsV[0][0][0]), 32, 64, SL, /*16DW*/3, /*8DW*/7);
  }
#else
  stage_copy(0, 0);
#endif

  for (int kt = 0; kt < NT; ++kt) {
    int buf = kt & 1;
    __syncthreads();                    // all waves done with buf[(kt+1)&1]
#if TDM_MODE
    if (w == 0) {
      if (kt + 1 < NT) {
        int nb = (kt + 1) & 1;
        tdm_load_2d(kbase + (size_t)(kt + 1) * 32 * DH, lds_off_of(&ldsK[nb][0][0]),
                    64, 32, DH, 4, 7);
        tdm_load_2d(vbase + (kt + 1) * 32, lds_off_of(&ldsV[nb][0][0]),
                    32, 64, SL, 3, 7);
        __builtin_amdgcn_s_wait_tensorcnt(2);   // tile kt done; kt+1 in flight
      } else {
        __builtin_amdgcn_s_wait_tensorcnt(0);
      }
    }
#else
    if (kt + 1 < NT) {
      stage_copy(kt + 1, (kt + 1) & 1);
      __builtin_prefetch(kbase + (size_t)(kt + 2) * 32 * DH + (t >> 3) * DH, 0, 1);
    }
#endif
    __syncthreads();                    // buf[kt&1] ready

    // Scores: 16 q-rows x 32 keys (2 n-subtiles, K-depth 64 over d)
    float sc[2][8];
#pragma unroll
    for (int nt = 0; nt < 2; ++nt) {
      int col = nt * 16 + lrow;         // key index within tile
      v16bf bk0 = *(const v16bf*)&ldsK[buf][col][16 * half];
      v16bf bk1 = *(const v16bf*)&ldsK[buf][col][32 + 16 * half];
      v8f s = {};
      s = wmma_bf16(qa, bk0, s);
      s = wmma_bf16(qb, bk1, s);
      int km = mask[b * SL + kt * 32 + col];
#pragma unroll
      for (int r = 0; r < 8; ++r) {
        float v = s[r] * alpha + beta;  // GhostBatchNorm affine (eval mode)
        if (km)       v = NEG_INF;      // masked key position
        if (rmask[r]) v = NEG_INF;      // masked query row
        sc[nt][r] = v;
      }
    }

    // Online softmax update (row reductions across 16-lane groups)
#pragma unroll
    for (int r = 0; r < 8; ++r) {
      float mx = fmaxf(sc[0][r], sc[1][r]);
#pragma unroll
      for (int d = 8; d >= 1; d >>= 1) mx = fmaxf(mx, __shfl_xor(mx, d, 32));
      float mn = fmaxf(m8[r], mx);
      float scale = __expf(m8[r] - mn);
      float p0 = __expf(sc[0][r] - mn);
      float p1 = __expf(sc[1][r] - mn);
      float rs = p0 + p1;
#pragma unroll
      for (int d = 8; d >= 1; d >>= 1) rs += __shfl_xor(rs, d, 32);
      l8[r] = l8[r] * scale + rs;
      m8[r] = mn;
#pragma unroll
      for (int nt2 = 0; nt2 < 4; ++nt2) o[nt2][r] *= scale;
      sc[0][r] = p0; sc[1][r] = p1;
    }

    // C-layout -> A-layout for P via per-wave LDS round trip (bf16)
#pragma unroll
    for (int nt = 0; nt < 2; ++nt) {
      int col = nt * 16 + lrow;
#pragma unroll
      for (int r = 0; r < 8; ++r)
        ldsP[w][r + 8 * half][col] = (bf16)sc[nt][r];
    }
    __syncthreads();

    V16U up;
    up.h[0] = *(const v8bf*)&ldsP[w][lrow][16 * half];
    up.h[1] = *(const v8bf*)&ldsP[w][lrow][16 * half + 8];
    v16bf pa = up.v;

    // O += P(16x32) * V(32x64)
#pragma unroll
    for (int nt2 = 0; nt2 < 4; ++nt2) {
      int col = nt2 * 16 + lrow;        // dv index
      v16bf bv = *(const v16bf*)&ldsV[buf][col][16 * half];
      o[nt2] = wmma_bf16(pa, bv, o[nt2]);
    }
  }

  // Epilogue: divide by running sum, zero masked query rows, coalesced store
#pragma unroll
  for (int nt2 = 0; nt2 < 4; ++nt2) {
    int dv = nt2 * 16 + lrow;
    float rv[8];
#pragma unroll
    for (int r = 0; r < 8; ++r)
      rv[r] = (rmask[r] || l8[r] <= 0.f) ? 0.f : o[nt2][r] / l8[r];
    float* base = out + ((size_t)(b * DA) + h * DH + dv) * SL + qrow0 + w * 16 + 8 * half;
    *(float4*)base       = make_float4(rv[0], rv[1], rv[2], rv[3]);
    *((float4*)base + 1) = make_float4(rv[4], rv[5], rv[6], rv[7]);
  }
}

// ---------------------------------------------------------------------------
// Host-side launcher
// ---------------------------------------------------------------------------
extern "C" void kernel_launch(void* const* d_in, const int* in_sizes, int n_in,
                              void* d_out, int out_size, void* d_ws, size_t ws_size,
                              hipStream_t stream) {
  const float* q    = (const float*)d_in[0];
  const float* k_in = (const float*)d_in[1];
  const float* v_in = (const float*)d_in[2];
  const float* k_w  = (const float*)d_in[3];
  const float* k_b  = (const float*)d_in[4];
  const float* v_w  = (const float*)d_in[5];
  const float* v_b  = (const float*)d_in[6];
  const float* gg   = (const float*)d_in[7];
  const float* gb   = (const float*)d_in[8];
  const float* gm   = (const float*)d_in[9];
  const float* gs   = (const float*)d_in[10];
  const int*   mask = (const int*)d_in[11];
  float* out = (float*)d_out;

  char* ws = (char*)d_ws;
  size_t off = 0;
  auto carve = [&](size_t bytes) {
    void* p = ws + off;
    off += (bytes + 255) & ~(size_t)255;
    return p;
  };
  bf16* qp   = (bf16*)carve((size_t)BS * H * SL * DH * 2);   // packed Q   [b,h,s,d]
  bf16* kp   = (bf16*)carve((size_t)BS * H * SL * DH * 2);   // packed K   [b,h,s,d]
  bf16* vbf  = (bf16*)carve((size_t)BS * DA * SL * 2);       // V          [b,c,s]
  bf16* kinT = (bf16*)carve((size_t)BS * SL * DA * 2);       // k_in^T     [b,s,c]
  bf16* vinT = (bf16*)carve((size_t)BS * SL * DA * 2);       // v_in^T     [b,s,c]
  bf16* kwb  = (bf16*)carve((size_t)DA * DA * 2);
  bf16* vwb  = (bf16*)carve((size_t)DA * DA * 2);

  const int nw4 = DA * DA / 4;
  cvt_f32_bf16<<<nw4 / 256, 256, 0, stream>>>(k_w, kwb, nw4);
  cvt_f32_bf16<<<nw4 / 256, 256, 0, stream>>>(v_w, vwb, nw4);
  pack_x<<<dim3(SL / 16, DA / 16, BS), dim3(16, 16), 0, stream>>>(k_in, kinT);
  pack_x<<<dim3(SL / 16, DA / 16, BS), dim3(16, 16), 0, stream>>>(v_in, vinT);
  pack_q<<<dim3(SL / 16, DH / 16, BS * H), dim3(16, 16), 0, stream>>>(q, qp);

  project_gemm<<<dim3(SL / 64, DA / 128, BS), 256, 0, stream>>>(kwb, k_b, kinT, kp, vbf, 0);
  project_gemm<<<dim3(SL / 64, DA / 128, BS), 256, 0, stream>>>(vwb, v_b, vinT, kp, vbf, 1);

  attention<<<dim3(SL / 128, H, BS), 256, 0, stream>>>(qp, kp, vbf, mask,
                                                       gg, gb, gm, gs, out);
}
